// HierarchicalEncoder_19215683682484
// MI455X (gfx1250) — matrix operand order repaired and checked by reading, not compiled, not run on hardware
//
#include <hip/hip_runtime.h>

// ---------------------------------------------------------------------------
// HierarchicalEncoder for MI455X (gfx1250, wave32, WMMA).
// Heavy GEMMs (edge message MLP: [2M,135]x[135,64] x3 layers ~104 GFLOPs, and
// node update [131072,128]x[128,64] x3) run on v_wmma_f32_16x16x32_f16.
// Gathers/scatters stay in L2 (h is 17-33MB, L2 = 192MB).
// ---------------------------------------------------------------------------

typedef __attribute__((ext_vector_type(16))) _Float16 v16h;
typedef __attribute__((ext_vector_type(8)))  float    v8f;

constexpr int kB  = 512;
constexpr int kN  = 131072;
constexpr int kE  = 2097152;
constexpr int kH  = 64;
constexpr int kNF = 4;
constexpr int kEF = 7;
// message MLP K: 64 (h_src) + 64 (h_dst) + 7 (edge_attr) padded to 160
constexpr int kKM = 160;
// update K: 64 (h) + 64 (agg/deg) = 128
constexpr int kKU = 128;

__device__ __forceinline__ unsigned pack2h(float a, float b) {
  union { _Float16 h[2]; unsigned u; } x;
  x.h[0] = (_Float16)a; x.h[1] = (_Float16)b;
  return x.u;
}

union AB16 { v16h v; uint4 u[2]; };

// ---------------------------------------------------------------------------
// Weight preprocessing: f32 -> f16, transposed to [n][k] so each WMMA B-lane
// reads two runs of 8 contiguous halves, with zero K-padding for the message MLP.
// ---------------------------------------------------------------------------
struct WSrc {
  const float* msgW[3];
  const float* selfW[3];
  const float* aggW[3];
};

__global__ __launch_bounds__(256) void k_convw(WSrc ws, _Float16* __restrict__ msgWT,
                                               _Float16* __restrict__ updWT) {
  int idx = blockIdx.x * 256 + threadIdx.x;
  const int msgTot = 3 * kH * kKM;   // 30720
  const int updTot = 3 * kH * kKU;   // 24576
  if (idx < msgTot) {
    int l   = idx / (kH * kKM);
    int rem = idx - l * (kH * kKM);
    int n   = rem / kKM;
    int k   = rem - n * kKM;
    float v = (k < 2 * kH + kEF) ? ws.msgW[l][k * kH + n] : 0.0f;
    msgWT[idx] = (_Float16)v;
  } else if (idx < msgTot + updTot) {
    int i2  = idx - msgTot;
    int l   = i2 / (kH * kKU);
    int rem = i2 - l * (kH * kKU);
    int n   = rem / kKU;
    int k   = rem - n * kKU;
    float v = (k < kH) ? ws.selfW[l][k * kH + n] : ws.aggW[l][(k - kH) * kH + n];
    updWT[i2] = (_Float16)v;
  }
}

// ---------------------------------------------------------------------------
// Edge stats: deg[dst] += 1 and branch-2 per-graph edge_attr sums + counts.
// LDS-staged per block to cut global atomic contention on 512 counters.
// ---------------------------------------------------------------------------
__global__ __launch_bounds__(256) void k_edge_stats(const int* __restrict__ ei,
                                                    const int* __restrict__ batch,
                                                    const float* __restrict__ ea,
                                                    float* __restrict__ deg,
                                                    float* __restrict__ evbuf) {
  __shared__ float evloc[kB * 8];
  int t = threadIdx.x;
  for (int j = 0; j < 16; ++j) evloc[t + j * 256] = 0.0f;
  __syncthreads();
  int base = blockIdx.x * (kE / 512);  // 4096 edges per block
  for (int i = 0; i < 16; ++i) {
    int e = base + i * 256 + t;
    int s = ei[e];
    int d = ei[kE + e];
    atomicAdd(deg + d, 1.0f);
    int g = batch[s];
    atomicAdd(&evloc[g * 8 + 7], 1.0f);
    const float* pe = ea + (size_t)e * kEF;
    for (int k = 0; k < kEF; ++k) atomicAdd(&evloc[g * 8 + k], pe[k]);
  }
  __syncthreads();
  for (int j = 0; j < 16; ++j) {
    int idx = t + j * 256;
    float v = evloc[idx];
    if (v != 0.0f) atomicAdd(evbuf + idx, v);
  }
}

// ---------------------------------------------------------------------------
// Encoder: h = relu(x @ enc_W + enc_b), K=4 -> plain VALU, coalesced.
// ---------------------------------------------------------------------------
__global__ __launch_bounds__(256) void k_encoder(const float* __restrict__ x,
                                                 const float* __restrict__ encW,
                                                 const float* __restrict__ encb,
                                                 float* __restrict__ h,
                                                 _Float16* __restrict__ h16) {
  int idx  = blockIdx.x * 256 + threadIdx.x;  // over N*H
  int node = idx >> 6;
  int d    = idx & 63;
  const float* xr = x + (size_t)node * kNF;
  float acc = encb[d];
#pragma unroll
  for (int k = 0; k < kNF; ++k) acc += xr[k] * encW[k * kH + d];
  acc = fmaxf(acc, 0.0f);
  h[idx]   = acc;
  h16[idx] = (_Float16)acc;
}

// ---------------------------------------------------------------------------
// Message kernel: one wave per 16-edge tile.
// A = [h16[src] | h16[dst] | ea | 0pad] staged in LDS (16 x 160 halves),
// C[16edges,16cols] x 4 col-tiles, K swept in 5 chunks of 32 -> 20 WMMAs/wave.
// Fused bias+ReLU, scatter atomicAdd into agg (sum; /deg applied later).
// ---------------------------------------------------------------------------
__global__ __launch_bounds__(256) void k_message(const _Float16* __restrict__ h16,
                                                 const float* __restrict__ ea,
                                                 const int* __restrict__ ei,
                                                 const _Float16* __restrict__ WT,
                                                 const float* __restrict__ mb,
                                                 float* __restrict__ agg) {
  __shared__ __align__(16) unsigned short A[8][16 * kKM];  // 40 KB
  __shared__ int DST[8][16];
  const int w = threadIdx.x >> 5;
  const int l = threadIdx.x & 31;
  const int tile = blockIdx.x * 8 + w;

  // ---- stage: 2 lanes per edge-row ----
  const int r  = l >> 1;
  const int hf = l & 1;
  const int e  = tile * 16 + r;
  const int s  = ei[e];
  const int d  = ei[kE + e];
  if (hf == 0) DST[w][r] = d;
  const uint4* ps = (const uint4*)(h16 + (size_t)s * kH);
  const uint4* pd = (const uint4*)(h16 + (size_t)d * kH);
  uint4* Ar = (uint4*)(&A[w][r * kKM]);  // 20 uint4 per row
#pragma unroll
  for (int j = 0; j < 4; ++j) Ar[hf * 4 + j] = ps[hf * 4 + j];       // K 0..63
#pragma unroll
  for (int j = 0; j < 4; ++j) Ar[8 + hf * 4 + j] = pd[hf * 4 + j];   // K 64..127
  if (hf == 0) {                                                      // K 128..159
    const float* pe = ea + (size_t)e * kEF;
    uint4 t0;
    t0.x = pack2h(pe[0], pe[1]);
    t0.y = pack2h(pe[2], pe[3]);
    t0.z = pack2h(pe[4], pe[5]);
    t0.w = pack2h(pe[6], 0.0f);
    uint4 z; z.x = z.y = z.z = z.w = 0u;
    Ar[16] = t0; Ar[17] = z; Ar[18] = z; Ar[19] = z;
  }
  __syncthreads();

  // ---- compute ----
  const int lh = l >> 4;   // lane-half selects K sub-blocks / M rows 8..15
  const int ln = l & 15;
  const uint4* Au = (const uint4*)(&A[w][0]);
#pragma unroll
  for (int nt = 0; nt < 4; ++nt) {
    const int n = nt * 16 + ln;
    const float bias = mb[n];
    v8f c = {0.f, 0.f, 0.f, 0.f, 0.f, 0.f, 0.f, 0.f};
    const uint4* Bu = (const uint4*)(WT + (size_t)n * kKM);
#pragma unroll
    for (int kc = 0; kc < 5; ++kc) {
      AB16 a, b;
      // A lane layout (16b, 16x32): K = 32kc + 8*lh + {0..7, 16..23}
      int ai = ln * 20 + kc * 4 + lh;
      a.u[0] = Au[ai];
      a.u[1] = Au[ai + 2];
      // B lane layout (16b, 32x16): K = 32kc + 8*lh + {0..7, 16..23}
      int bi = kc * 4 + lh;
      b.u[0] = Bu[bi];
      b.u[1] = Bu[bi + 2];
      c = __builtin_amdgcn_wmma_f32_16x16x32_f16(false, a.v, false, b.v,
                                                 (short)0, c, false, false);
    }
#pragma unroll
    for (int rr = 0; rr < 8; ++rr) {
      int row = rr + 8 * lh;               // C layout: VGPR rr -> M = rr + 8*lh
      float m = fmaxf(c[rr] + bias, 0.0f); // relu(msg)
      int dn  = DST[w][row];
      atomicAdd(agg + (size_t)dn * kH + n, m);
    }
  }
}

// ---------------------------------------------------------------------------
// Update kernel: h = relu(h@selfW + (agg/deg)@aggW + b) + h, in place.
// One wave per 16-node tile, K=128, 16 WMMAs/wave.
// ---------------------------------------------------------------------------
__global__ __launch_bounds__(256) void k_update(float* __restrict__ h,
                                                _Float16* __restrict__ h16,
                                                const float* __restrict__ agg,
                                                const float* __restrict__ deg,
                                                const _Float16* __restrict__ WT,
                                                const float* __restrict__ ub) {
  __shared__ __align__(16) unsigned short A[8][16 * kKU];  // 32 KB
  const int w = threadIdx.x >> 5;
  const int l = threadIdx.x & 31;
  const int tile = blockIdx.x * 8 + w;

  const int r    = l >> 1;
  const int hf   = l & 1;
  const int node = tile * 16 + r;
  uint4* Ar = (uint4*)(&A[w][r * kKU]);  // 16 uint4 per row
  const uint4* ph = (const uint4*)(h16 + (size_t)node * kH);
#pragma unroll
  for (int j = 0; j < 4; ++j) Ar[hf * 4 + j] = ph[hf * 4 + j];  // K 0..63
  const float invd = 1.0f / fmaxf(deg[node], 1.0f);
  const float4* pa = (const float4*)(agg + (size_t)node * kH + hf * 32);
#pragma unroll
  for (int j = 0; j < 4; ++j) {                                  // K 64..127
    float4 f0 = pa[2 * j];
    float4 f1 = pa[2 * j + 1];
    uint4 o;
    o.x = pack2h(f0.x * invd, f0.y * invd);
    o.y = pack2h(f0.z * invd, f0.w * invd);
    o.z = pack2h(f1.x * invd, f1.y * invd);
    o.w = pack2h(f1.z * invd, f1.w * invd);
    Ar[8 + hf * 4 + j] = o;
  }
  __syncthreads();

  const int lh = l >> 4;
  const int ln = l & 15;
  const uint4* Au = (const uint4*)(&A[w][0]);
#pragma unroll
  for (int nt = 0; nt < 4; ++nt) {
    const int n = nt * 16 + ln;
    const float bias = ub[n];
    v8f c = {0.f, 0.f, 0.f, 0.f, 0.f, 0.f, 0.f, 0.f};
    const uint4* Bu = (const uint4*)(WT + (size_t)n * kKU);
#pragma unroll
    for (int kc = 0; kc < 4; ++kc) {
      AB16 a, b;
      int ai = ln * 16 + kc * 4 + lh;
      a.u[0] = Au[ai];
      a.u[1] = Au[ai + 2];
      int bi = kc * 4 + lh;
      b.u[0] = Bu[bi];
      b.u[1] = Bu[bi + 2];
      c = __builtin_amdgcn_wmma_f32_16x16x32_f16(false, a.v, false, b.v,
                                                 (short)0, c, false, false);
    }
#pragma unroll
    for (int rr = 0; rr < 8; ++rr) {
      int nd = tile * 16 + rr + 8 * lh;
      size_t off = (size_t)nd * kH + n;
      float v = fmaxf(c[rr] + bias, 0.0f) + h[off];  // relu + residual
      h[off]   = v;
      h16[off] = (_Float16)v;
    }
  }
}

// ---------------------------------------------------------------------------
// Branch 1 pooling: batch is repeat(arange(B), 256) -> block per graph.
// ---------------------------------------------------------------------------
__global__ __launch_bounds__(256) void k_pool(const float* __restrict__ h,
                                              float* __restrict__ gmean,
                                              float* __restrict__ gmax) {
  __shared__ float S[4][64], M[4][64];
  int g = blockIdx.x;
  int t = threadIdx.x;
  int d = t & 63;
  int q = t >> 6;
  float s = 0.0f, mx = -3.402823466e38f;
  int base = g * 256 + q * 64;
  for (int i = 0; i < 64; ++i) {
    float v = h[(size_t)(base + i) * kH + d];
    s += v;
    mx = fmaxf(mx, v);
  }
  S[q][d] = s; M[q][d] = mx;
  __syncthreads();
  if (q == 0) {
    s  = S[0][d] + S[1][d] + S[2][d] + S[3][d];
    mx = fmaxf(fmaxf(M[0][d], M[1][d]), fmaxf(M[2][d], M[3][d]));
    gmean[g * kH + d] = s * (1.0f / 256.0f);
    gmax[g * kH + d]  = mx;
  }
}

// ---------------------------------------------------------------------------
// Head MLPs: block per graph (tiny). out layout: z[0,4096) mu[4096,8192) lv[8192,12288)
// ---------------------------------------------------------------------------
__global__ __launch_bounds__(64) void k_topo(const float* __restrict__ gmean,
                                             const float* __restrict__ gmax,
                                             const float* __restrict__ W1, const float* __restrict__ b1,
                                             const float* __restrict__ W2, const float* __restrict__ b2,
                                             const float* __restrict__ Wm, const float* __restrict__ bm,
                                             const float* __restrict__ Wv, const float* __restrict__ bv,
                                             float* __restrict__ out) {
  __shared__ float in[128], h1[64], h2[32];
  int g = blockIdx.x, t = threadIdx.x;
  in[t]      = gmean[g * kH + t];
  in[64 + t] = gmax[g * kH + t];
  __syncthreads();
  float acc = b1[t];
  for (int k = 0; k < 128; ++k) acc += in[k] * W1[k * 64 + t];
  h1[t] = fmaxf(acc, 0.0f);
  __syncthreads();
  if (t < 32) {
    acc = b2[t];
    for (int k = 0; k < 64; ++k) acc += h1[k] * W2[k * 32 + t];
    h2[t] = fmaxf(acc, 0.0f);
  }
  __syncthreads();
  if (t < 2) {
    float mu = bm[t], lv = bv[t];
    for (int k = 0; k < 32; ++k) { mu += h2[k] * Wm[k * 2 + t]; lv += h2[k] * Wv[k * 2 + t]; }
    out[g * 8 + t]            = mu;  // z
    out[4096 + g * 8 + t]     = mu;  // mu
    out[8192 + g * 8 + t]     = lv;  // logvar
  }
}

__global__ __launch_bounds__(64) void k_val(const float* __restrict__ evbuf,
                                            const float* __restrict__ W1, const float* __restrict__ b1,
                                            const float* __restrict__ W2, const float* __restrict__ b2,
                                            const float* __restrict__ Wm, const float* __restrict__ bm,
                                            const float* __restrict__ Wv, const float* __restrict__ bv,
                                            float* __restrict__ out) {
  __shared__ float in[7], h1[64], h2[32];
  int g = blockIdx.x, t = threadIdx.x;
  if (t < 7) {
    float cnt = evbuf[g * 8 + 7];
    in[t] = (cnt > 0.0f) ? evbuf[g * 8 + t] / fmaxf(cnt, 1.0f) : 0.0f;
  }
  __syncthreads();
  float acc = b1[t];
  for (int k = 0; k < 7; ++k) acc += in[k] * W1[k * 64 + t];
  h1[t] = fmaxf(acc, 0.0f);
  __syncthreads();
  if (t < 32) {
    acc = b2[t];
    for (int k = 0; k < 64; ++k) acc += h1[k] * W2[k * 32 + t];
    h2[t] = fmaxf(acc, 0.0f);
  }
  __syncthreads();
  if (t < 2) {
    float mu = bm[t], lv = bv[t];
    for (int k = 0; k < 32; ++k) { mu += h2[k] * Wm[k * 2 + t]; lv += h2[k] * Wv[k * 2 + t]; }
    out[g * 8 + 2 + t]        = mu;
    out[4096 + g * 8 + 2 + t] = mu;
    out[8192 + g * 8 + 2 + t] = lv;
  }
}

__global__ __launch_bounds__(64) void k_pz(const float* __restrict__ poles, const float* __restrict__ zrs,
                                           const int* __restrict__ pcnt, const int* __restrict__ zcnt,
                                           const float* __restrict__ pW1, const float* __restrict__ pb1,
                                           const float* __restrict__ pW2, const float* __restrict__ pb2,
                                           const float* __restrict__ pWr, const float* __restrict__ pbr,
                                           const float* __restrict__ zW1, const float* __restrict__ zb1,
                                           const float* __restrict__ zW2, const float* __restrict__ zb2,
                                           const float* __restrict__ zWr, const float* __restrict__ zbr,
                                           const float* __restrict__ cW, const float* __restrict__ cb,
                                           const float* __restrict__ Wm, const float* __restrict__ bm,
                                           const float* __restrict__ Wv, const float* __restrict__ bv,
                                           float* __restrict__ out) {
  __shared__ float phi1[2][32], pool[2][32], rho[32], comb[32];
  int g = blockIdx.x, t = threadIdx.x;
  int u = t & 31, st = t >> 5;  // wave0 = poles, wave1 = zeros
  const float* S  = st ? zrs : poles;
  int cnt         = st ? zcnt[g] : pcnt[g];
  const float* W1 = st ? zW1 : pW1;  const float* b1 = st ? zb1 : pb1;
  const float* W2 = st ? zW2 : pW2;  const float* b2 = st ? zb2 : pb2;
  const float* Wr = st ? zWr : pWr;  const float* br = st ? zbr : pbr;
  float pooled = 0.0f;
  for (int p = 0; p < 8; ++p) {
    float s0 = S[(g * 8 + p) * 2 + 0];
    float s1 = S[(g * 8 + p) * 2 + 1];
    phi1[st][u] = fmaxf(W1[u] * s0 + W1[32 + u] * s1 + b1[u], 0.0f);
    __syncthreads();
    float acc = b2[u];
    for (int k = 0; k < 32; ++k) acc += phi1[st][k] * W2[k * 32 + u];
    if (p < cnt) pooled += fmaxf(acc, 0.0f);
    __syncthreads();
  }
  pool[st][u] = pooled;
  __syncthreads();
  if (u < 16) {
    float acc = br[u];
    for (int k = 0; k < 32; ++k) acc += pool[st][k] * Wr[k * 16 + u];
    rho[st * 16 + u] = (cnt > 0) ? acc : 0.0f;
  }
  __syncthreads();
  if (t < 32) {
    float acc = cb[t];
    for (int k = 0; k < 32; ++k) acc += rho[k] * cW[k * 32 + t];
    comb[t] = fmaxf(acc, 0.0f);
  }
  __syncthreads();
  if (t < 4) {
    float mu = bm[t], lv = bv[t];
    for (int k = 0; k < 32; ++k) { mu += comb[k] * Wm[k * 4 + t]; lv += comb[k] * Wv[k * 4 + t]; }
    out[g * 8 + 4 + t]        = mu;
    out[4096 + g * 8 + 4 + t] = mu;
    out[8192 + g * 8 + 4 + t] = lv;
  }
}

// ---------------------------------------------------------------------------
// Host launch
// ---------------------------------------------------------------------------
struct P {
  const float *x, *ea, *poles, *zeros, *encW, *encb;
  const float *msgW[3], *msgb[3], *selfW[3], *aggW[3], *updb[3];
  const float *t1W, *t1b, *t2W, *t2b, *tmuW, *tmub, *tlvW, *tlvb;
  const float *v1W, *v1b, *v2W, *v2b, *vmuW, *vmub, *vlvW, *vlvb;
  const float *pp1W, *pp1b, *pp2W, *pp2b, *prW, *prb;
  const float *zp1W, *zp1b, *zp2W, *zp2b, *zrW, *zrb;
  const float *combW, *combb, *pzmuW, *pzmub, *pzlvW, *pzlvb;
  const int *ei, *batch, *pcnt, *zcnt;
};

extern "C" void kernel_launch(void* const* d_in, const int* in_sizes, int n_in,
                              void* d_out, int out_size, void* d_ws, size_t ws_size,
                              hipStream_t stream) {
  auto F = [&](int i) { return (const float*)d_in[i]; };
  auto I = [&](int i) { return (const int*)d_in[i]; };
  P p;
  if (in_sizes[0] == kN * kNF) {
    // insertion-order flatten of setup_inputs (params nested in place)
    p.x = F(0); p.ea = F(1); p.poles = F(2); p.zeros = F(3);
    p.encW = F(4); p.encb = F(5);
    for (int l = 0; l < 3; ++l) {
      int b = 6 + 5 * l;
      p.msgW[l] = F(b); p.msgb[l] = F(b + 1); p.selfW[l] = F(b + 2);
      p.aggW[l] = F(b + 3); p.updb[l] = F(b + 4);
    }
    p.t1W = F(21); p.t1b = F(22); p.t2W = F(23); p.t2b = F(24);
    p.tmuW = F(25); p.tmub = F(26); p.tlvW = F(27); p.tlvb = F(28);
    p.v1W = F(29); p.v1b = F(30); p.v2W = F(31); p.v2b = F(32);
    p.vmuW = F(33); p.vmub = F(34); p.vlvW = F(35); p.vlvb = F(36);
    p.pp1W = F(37); p.pp1b = F(38); p.pp2W = F(39); p.pp2b = F(40);
    p.prW = F(41); p.prb = F(42);
    p.zp1W = F(43); p.zp1b = F(44); p.zp2W = F(45); p.zp2b = F(46);
    p.zrW = F(47); p.zrb = F(48);
    p.combW = F(49); p.combb = F(50);
    p.pzmuW = F(51); p.pzmub = F(52); p.pzlvW = F(53); p.pzlvb = F(54);
    p.ei = I(55); p.batch = I(56); p.pcnt = I(57); p.zcnt = I(58);
  } else {
    // jax sorted-key pytree flatten
    p.batch = I(0); p.ea = F(1); p.ei = I(2);
    p.combW = F(3); p.combb = F(4); p.encW = F(5); p.encb = F(6);
    for (int l = 0; l < 3; ++l) {
      int b = 7 + 5 * l;
      p.aggW[l] = F(b); p.msgW[l] = F(b + 1); p.msgb[l] = F(b + 2);
      p.selfW[l] = F(b + 3); p.updb[l] = F(b + 4);
    }
    p.pp1W = F(22); p.pp1b = F(23); p.pp2W = F(24); p.pp2b = F(25);
    p.prW = F(26); p.prb = F(27);
    p.pzlvW = F(28); p.pzlvb = F(29); p.pzmuW = F(30); p.pzmub = F(31);
    p.tlvW = F(32); p.tlvb = F(33); p.tmuW = F(34); p.tmub = F(35);
    p.t1W = F(36); p.t1b = F(37); p.t2W = F(38); p.t2b = F(39);
    p.v1W = F(40); p.v1b = F(41); p.v2W = F(42); p.v2b = F(43);
    p.vlvW = F(44); p.vlvb = F(45); p.vmuW = F(46); p.vmub = F(47);
    p.zp1W = F(48); p.zp1b = F(49); p.zp2W = F(50); p.zp2b = F(51);
    p.zrW = F(52); p.zrb = F(53);
    p.pcnt = I(54); p.poles = F(55); p.x = F(56); p.zcnt = I(57); p.zeros = F(58);
  }

  // workspace carve (~85 MB)
  size_t off = 0;
  auto carve = [&](size_t bytes) -> void* {
    off = (off + 255) & ~(size_t)255;
    void* q = (char*)d_ws + off;
    off += bytes;
    return q;
  };
  float*    h       = (float*)carve((size_t)kN * kH * 4);
  _Float16* h16     = (_Float16*)carve((size_t)kN * kH * 2);
  float*    agg     = (float*)carve((size_t)kN * kH * 4);
  float*    deg     = (float*)carve((size_t)kN * 4);
  float*    evbuf   = (float*)carve((size_t)kB * 8 * 4);
  float*    gmean   = (float*)carve((size_t)kB * kH * 4);
  float*    gmax    = (float*)carve((size_t)kB * kH * 4);
  _Float16* msgWT16 = (_Float16*)carve((size_t)3 * kH * kKM * 2);
  _Float16* updWT16 = (_Float16*)carve((size_t)3 * kH * kKU * 2);
  (void)ws_size; (void)n_in; (void)out_size;

  float* out = (float*)d_out;

  hipMemsetAsync(deg, 0, (size_t)kN * 4, stream);
  hipMemsetAsync(evbuf, 0, (size_t)kB * 8 * 4, stream);

  WSrc wsrc;
  for (int l = 0; l < 3; ++l) { wsrc.msgW[l] = p.msgW[l]; wsrc.selfW[l] = p.selfW[l]; wsrc.aggW[l] = p.aggW[l]; }
  k_convw<<<216, 256, 0, stream>>>(wsrc, msgWT16, updWT16);

  k_edge_stats<<<512, 256, 0, stream>>>(p.ei, p.batch, p.ea, deg, evbuf);
  k_encoder<<<(kN * kH) / 256, 256, 0, stream>>>(p.x, p.encW, p.encb, h, h16);

  for (int l = 0; l < 3; ++l) {
    hipMemsetAsync(agg, 0, (size_t)kN * kH * 4, stream);
    k_message<<<kE / 128, 256, 0, stream>>>(h16, p.ea, p.ei,
                                            msgWT16 + (size_t)l * kH * kKM,
                                            p.msgb[l], agg);
    k_update<<<kN / 128, 256, 0, stream>>>(h, h16, agg, deg,
                                           updWT16 + (size_t)l * kH * kKU,
                                           p.updb[l]);
  }

  k_pool<<<kB, 256, 0, stream>>>(h, gmean, gmax);
  k_topo<<<kB, 64, 0, stream>>>(gmean, gmax, p.t1W, p.t1b, p.t2W, p.t2b,
                                p.tmuW, p.tmub, p.tlvW, p.tlvb, out);
  k_val<<<kB, 64, 0, stream>>>(evbuf, p.v1W, p.v1b, p.v2W, p.v2b,
                               p.vmuW, p.vmub, p.vlvW, p.vlvb, out);
  k_pz<<<kB, 64, 0, stream>>>(p.poles, p.zeros, p.pcnt, p.zcnt,
                              p.pp1W, p.pp1b, p.pp2W, p.pp2b, p.prW, p.prb,
                              p.zp1W, p.zp1b, p.zp2W, p.zp2b, p.zrW, p.zrb,
                              p.combW, p.combb, p.pzmuW, p.pzmub, p.pzlvW, p.pzlvb,
                              out);
}
